// NNCLR_Embedding_40424232190018
// MI455X (gfx1250) — matrix operand order repaired and checked by reading, not compiled, not run on hardware
//
#include <hip/hip_runtime.h>

// ---------------- types ----------------
typedef __bf16 bf16;
typedef __attribute__((ext_vector_type(16))) bf16  v16bf;
typedef __attribute__((ext_vector_type(8)))  bf16  v8bf;
typedef __attribute__((ext_vector_type(8)))  float v8f;

static __device__ inline v8f wmma_bf16(v16bf a, v16bf b, v8f c) {
    // D = A(16x32 bf16) * B(32x16 bf16) + C(16x16 f32)
    return __builtin_amdgcn_wmma_f32_16x16x32_bf16(
        /*neg_a=*/false, a, /*neg_b=*/false, b,
        /*c_mod=*/(short)0, c, /*reuse_a=*/false, /*reuse_b=*/false);
}

#define NEG_BIG (-3.0e38f)

// ---------------- small utility kernels ----------------
__global__ __launch_bounds__(256) void cvt_f32_to_bf16_kernel(
    const float* __restrict__ src, bf16* __restrict__ dst, size_t n) {
    size_t i = (size_t)blockIdx.x * blockDim.x + threadIdx.x;
    size_t stride = (size_t)gridDim.x * blockDim.x;
    for (; i < n; i += stride) dst[i] = (bf16)src[i];
}

// dst[j*D+k] = (bf16) src[k*D+j]   (transpose so GEMMs become A * B^T)
__global__ __launch_bounds__(256) void transpose_cvt_kernel(
    const float* __restrict__ src, bf16* __restrict__ dst, int D) {
    int j = blockIdx.x;          // output row (original column)
    int k = threadIdx.x;         // output col (original row)
    dst[(size_t)j * D + k] = (bf16)src[(size_t)k * D + j];
}

__global__ __launch_bounds__(256) void copy_f32_kernel(
    const float* __restrict__ src, float* __restrict__ dst, size_t n) {
    size_t i = (size_t)blockIdx.x * blockDim.x + threadIdx.x;
    size_t stride = (size_t)gridDim.x * blockDim.x;
    for (; i < n; i += stride) dst[i] = src[i];
}

__global__ __launch_bounds__(256) void gather_rows_kernel(
    const bf16* __restrict__ queue, const int* __restrict__ idx,
    bf16* __restrict__ dst) {
    int i = blockIdx.x;
    int k = threadIdx.x;
    dst[(size_t)i * 256 + k] = queue[(size_t)idx[i] * 256 + k];
}

// ---------------- L2 normalize rows of z [rows x 256] ----------------
__global__ __launch_bounds__(256) void l2norm_kernel(
    const float* __restrict__ z, float* __restrict__ zn, bf16* __restrict__ znb) {
    __shared__ float red[8];
    int row = blockIdx.x, tid = threadIdx.x;
    float x = z[(size_t)row * 256 + tid];
    float ss = x * x;
#pragma unroll
    for (int o = 16; o >= 1; o >>= 1) ss += __shfl_xor(ss, o, 32);
    if ((tid & 31) == 0) red[tid >> 5] = ss;
    __syncthreads();
    float tot = 0.f;
#pragma unroll
    for (int i = 0; i < 8; i++) tot += red[i];
    float denom = fmaxf(sqrtf(tot), 1e-12f);
    float y = x / denom;
    zn[(size_t)row * 256 + tid] = y;
    znb[(size_t)row * 256 + tid] = (bf16)y;
}

// ---------------- WMMA GEMM: C[M,N] = act(A[M,256] * B[N,256]^T * scale + bias) --------
// Block: 16 rows (M-tile) x 128 cols (8 waves * 16). K = 256 fully unrolled.
__global__ __launch_bounds__(256) void gemm_bf16_kernel(
    const bf16* __restrict__ A, const bf16* __restrict__ B,
    const float* __restrict__ bias, float* __restrict__ Cf, bf16* __restrict__ Cb,
    int N, float scale, int relu) {
    __shared__ bf16 sA[16 * 256];
    int tid = threadIdx.x;
    int mbase = blockIdx.x * 16;
    int nbase = blockIdx.y * 128;
    // cooperative A tile load: 4096 bf16 = 16 per thread (32B each)
    *(v16bf*)&sA[tid * 16] = *(const v16bf*)(A + (size_t)mbase * 256 + (size_t)tid * 16);
    __syncthreads();

    int w = tid >> 5, lane = tid & 31;
    int n0 = lane & 15, h = lane >> 4;

    // Preload A fragments for all 8 K-steps (contiguous layout, matching B):
    // lane-half h supplies K = {16h..16h+15} inside each 32-chunk.
    v16bf afrag[8];
#pragma unroll
    for (int ks = 0; ks < 8; ks++) {
        int kc = ks * 32;
        union { v16bf v; v8bf p[2]; } ua;
        ua.p[0] = *(const v8bf*)&sA[n0 * 256 + kc + 16 * h];
        ua.p[1] = *(const v8bf*)&sA[n0 * 256 + kc + 16 * h + 8];
        afrag[ks] = ua.v;
    }

    int ncol = nbase + w * 16 + n0;
    const bf16* bp = B + (size_t)ncol * 256 + 16 * h;   // B layout: lane n=l&15, K=16h+e
    v8f acc = {0.f, 0.f, 0.f, 0.f, 0.f, 0.f, 0.f, 0.f};
#pragma unroll
    for (int ks = 0; ks < 8; ks++) {
        v16bf b = *(const v16bf*)(bp + ks * 32);
        acc = wmma_bf16(afrag[ks], b, acc);
    }

    float bv = bias ? bias[ncol] : 0.0f;
#pragma unroll
    for (int v = 0; v < 8; v++) {
        float x = acc[v] * scale + bv;            // D layout: row m=v+8h, col n
        if (relu) x = fmaxf(x, 0.0f);
        size_t o = (size_t)(mbase + v + 8 * h) * (size_t)N + (size_t)ncol;
        if (Cf) Cf[o] = x;
        if (Cb) Cb[o] = (bf16)x;
    }
}

// ---------------- fused NN search: argmax_q  zn[32 rows] . queue[q] ----------------
// M-tile = 32 (two register-resident A fragments) so every 1KB queue B-fragment
// feeds 2 WMMAs -> halves total L2 queue traffic vs a 16-row tile.
__global__ __launch_bounds__(256) void nn_search_kernel(
    const bf16* __restrict__ zn, const bf16* __restrict__ queue,
    int* __restrict__ out_idx, int Q) {
    __shared__ bf16 sA[32 * 256];
    __shared__ float sBest[8][32];
    __shared__ int   sIdx[8][32];
    int tid = threadIdx.x;
    int rowbase = blockIdx.x * 32;
    // cooperative A tile load: 8192 bf16 = 32 per thread
    *(v16bf*)&sA[tid * 16]        = *(const v16bf*)(zn + (size_t)rowbase * 256 + (size_t)tid * 16);
    *(v16bf*)&sA[4096 + tid * 16] = *(const v16bf*)(zn + (size_t)rowbase * 256 + 4096 + (size_t)tid * 16);
    __syncthreads();

    int w = tid >> 5, lane = tid & 31;
    int n0 = lane & 15, h = lane >> 4;

    v16bf afrag[2][8];
#pragma unroll
    for (int t = 0; t < 2; t++) {
#pragma unroll
        for (int ks = 0; ks < 8; ks++) {
            int kc = ks * 32;
            int base = (t * 16 + n0) * 256;
            union { v16bf v; v8bf p[2]; } ua;
            ua.p[0] = *(const v8bf*)&sA[base + kc + 16 * h];
            ua.p[1] = *(const v8bf*)&sA[base + kc + 16 * h + 8];
            afrag[t][ks] = ua.v;
        }
    }

    float best[2][8];
    int   bidx[2][8];
#pragma unroll
    for (int t = 0; t < 2; t++)
#pragma unroll
        for (int v = 0; v < 8; v++) { best[t][v] = NEG_BIG; bidx[t][v] = 0; }

    // each wave covers a distinct 16-column slice of every 128-wide Q stripe
    for (int qb = w * 16; qb < Q; qb += 128) {
        int q = qb + n0;                                  // this lane's queue row
        const bf16* bp = queue + (size_t)q * 256 + 16 * h;
        // prefetch next stripe for this wave (global_prefetch_b8)
        __builtin_prefetch(bp + (size_t)128 * 256, 0, 1);
        v8f acc0 = {0.f, 0.f, 0.f, 0.f, 0.f, 0.f, 0.f, 0.f};
        v8f acc1 = {0.f, 0.f, 0.f, 0.f, 0.f, 0.f, 0.f, 0.f};
#pragma unroll
        for (int ks = 0; ks < 8; ks++) {
            v16bf b = *(const v16bf*)(bp + ks * 32);
            acc0 = wmma_bf16(afrag[0][ks], b, acc0);
            acc1 = wmma_bf16(afrag[1][ks], b, acc1);
        }
#pragma unroll
        for (int v = 0; v < 8; v++) {
            float s0 = acc0[v];                           // sim(row v+8h,      queue q)
            float s1 = acc1[v];                           // sim(row 16+v+8h,   queue q)
            if (s0 > best[0][v]) { best[0][v] = s0; bidx[0][v] = q; }
            if (s1 > best[1][v]) { best[1][v] = s1; bidx[1][v] = q; }
        }
    }

    // reduce across the 16 columns (lane halves stay separate for masks 1..8)
#pragma unroll
    for (int o = 8; o >= 1; o >>= 1) {
#pragma unroll
        for (int t = 0; t < 2; t++)
#pragma unroll
            for (int v = 0; v < 8; v++) {
                float ov = __shfl_xor(best[t][v], o, 32);
                int   oi = __shfl_xor(bidx[t][v], o, 32);
                if (ov > best[t][v] || (ov == best[t][v] && oi < bidx[t][v])) {
                    best[t][v] = ov; bidx[t][v] = oi;
                }
            }
    }
    if (n0 == 0) {
#pragma unroll
        for (int t = 0; t < 2; t++)
#pragma unroll
            for (int v = 0; v < 8; v++) {
                sBest[w][t * 16 + v + 8 * h] = best[t][v];
                sIdx[w][t * 16 + v + 8 * h]  = bidx[t][v];
            }
    }
    __syncthreads();
    if (tid < 32) {
        float bb = sBest[0][tid]; int bi = sIdx[0][tid];
#pragma unroll
        for (int ww = 1; ww < 8; ww++) {
            float ov = sBest[ww][tid]; int oi = sIdx[ww][tid];
            if (ov > bb || (ov == bb && oi < bi)) { bb = ov; bi = oi; }
        }
        out_idx[rowbase + tid] = bi;
    }
}

// ---------------- row logsumexp (block per row) + diagonal grab ----------------
__global__ __launch_bounds__(256) void row_lse_kernel(
    const float* __restrict__ L, int N, float* __restrict__ lse, float* __restrict__ diag) {
    __shared__ float red[8];
    int row = blockIdx.x, tid = threadIdx.x;
    const float* p = L + (size_t)row * N;
    float mx = NEG_BIG;
    for (int j = tid; j < N; j += 256) mx = fmaxf(mx, p[j]);
#pragma unroll
    for (int o = 16; o >= 1; o >>= 1) mx = fmaxf(mx, __shfl_xor(mx, o, 32));
    if ((tid & 31) == 0) red[tid >> 5] = mx;
    __syncthreads();
    float M = red[0];
#pragma unroll
    for (int i = 1; i < 8; i++) M = fmaxf(M, red[i]);
    __syncthreads();
    float s = 0.f;
    for (int j = tid; j < N; j += 256) s += __expf(p[j] - M);
#pragma unroll
    for (int o = 16; o >= 1; o >>= 1) s += __shfl_xor(s, o, 32);
    if ((tid & 31) == 0) red[tid >> 5] = s;
    __syncthreads();
    if (tid == 0) {
        float S = 0.f;
        for (int i = 0; i < 8; i++) S += red[i];
        lse[row] = M + __logf(S);
        diag[row] = p[(size_t)row];   // p already offset by row*N -> element [row,row]
    }
}

// ---------------- column logsumexp: online partials over row chunks ----------------
__global__ __launch_bounds__(256) void col_lse_partial_kernel(
    const float* __restrict__ L, int N, int rowsPerChunk,
    float* __restrict__ pm, float* __restrict__ ps) {
    int col = blockIdx.x * 256 + threadIdx.x;
    int chunk = blockIdx.y;
    int j0 = chunk * rowsPerChunk;
    float m = NEG_BIG, s = 0.f;
    for (int j = j0; j < j0 + rowsPerChunk; j++) {
        float x = L[(size_t)j * N + col];
        float nm = fmaxf(m, x);
        s = s * __expf(m - nm) + __expf(x - nm);
        m = nm;
    }
    pm[(size_t)chunk * N + col] = m;
    ps[(size_t)chunk * N + col] = s;
}

__global__ __launch_bounds__(256) void col_lse_combine_kernel(
    const float* __restrict__ pm, const float* __restrict__ ps, int N, int chunks,
    float* __restrict__ lse) {
    int col = blockIdx.x * 256 + threadIdx.x;
    float m = NEG_BIG;
    for (int c = 0; c < chunks; c++) m = fmaxf(m, pm[(size_t)c * N + col]);
    float s = 0.f;
    for (int c = 0; c < chunks; c++) s += ps[(size_t)c * N + col] * __expf(pm[(size_t)c * N + col] - m);
    lse[col] = m + __logf(s);
}

// ---------------- final loss ----------------
__global__ __launch_bounds__(256) void final_loss_kernel(
    const float* __restrict__ lr1, const float* __restrict__ lc1, const float* __restrict__ dg1,
    const float* __restrict__ lr2, const float* __restrict__ lc2, const float* __restrict__ dg2,
    int Bn, float* __restrict__ out) {
    __shared__ float red[8];
    int tid = threadIdx.x;
    float s = 0.f;
    for (int i = tid; i < Bn; i += 256)
        s += lr1[i] + lc1[i] - 2.f * dg1[i] + lr2[i] + lc2[i] - 2.f * dg2[i];
#pragma unroll
    for (int o = 16; o >= 1; o >>= 1) s += __shfl_xor(s, o, 32);
    if ((tid & 31) == 0) red[tid >> 5] = s;
    __syncthreads();
    if (tid == 0) {
        float S = 0.f;
        for (int i = 0; i < 8; i++) S += red[i];
        out[0] = S / (4.0f * (float)Bn);
    }
}

// ---------------- driver ----------------
extern "C" void kernel_launch(void* const* d_in, const int* in_sizes, int n_in,
                              void* d_out, int out_size, void* d_ws, size_t ws_size,
                              hipStream_t stream) {
    (void)in_sizes; (void)n_in; (void)out_size; (void)ws_size;
    const float* h1 = (const float*)d_in[0];
    const float* h2 = (const float*)d_in[1];
    const float* W1 = (const float*)d_in[2];
    const float* b1 = (const float*)d_in[3];
    const float* W2 = (const float*)d_in[4];
    const float* b2 = (const float*)d_in[5];
    const float* fq = (const float*)d_in[6];
    float* out = (float*)d_out;

    const int Bsz = 2048, D = 256, Q = 65536, M2 = 4096;
    const float INV_TEMP = 10.0f;

    char* wsb = (char*)d_ws;
    size_t off = 0;
    auto alloc = [&](size_t bytes) -> char* {
        char* p = wsb + off;
        off = (off + bytes + 255) & ~(size_t)255;
        return p;
    };
    bf16* queue_b = (bf16*)alloc((size_t)Q * D * 2);     // 32 MB (L2-resident)
    bf16* h_b     = (bf16*)alloc((size_t)M2 * D * 2);
    bf16* w1t     = (bf16*)alloc((size_t)D * D * 2);
    bf16* w2t     = (bf16*)alloc((size_t)D * D * 2);
    bf16* y_b     = (bf16*)alloc((size_t)M2 * D * 2);
    float* z_f    = (float*)alloc((size_t)M2 * D * 4);
    float* zn_f   = (float*)alloc((size_t)M2 * D * 4);
    bf16* zn_b    = (bf16*)alloc((size_t)M2 * D * 2);
    int*  nn_idx  = (int*)alloc((size_t)M2 * 4);
    bf16* nn_b    = (bf16*)alloc((size_t)M2 * D * 2);
    float* L      = (float*)alloc((size_t)Bsz * Bsz * 4); // 16 MB, reused for L1 then L2
    float* lr1 = (float*)alloc((size_t)Bsz * 4);
    float* lc1 = (float*)alloc((size_t)Bsz * 4);
    float* dg1 = (float*)alloc((size_t)Bsz * 4);
    float* lr2 = (float*)alloc((size_t)Bsz * 4);
    float* lc2 = (float*)alloc((size_t)Bsz * 4);
    float* dg2 = (float*)alloc((size_t)Bsz * 4);
    float* pm  = (float*)alloc((size_t)8 * Bsz * 4);
    float* ps  = (float*)alloc((size_t)8 * Bsz * 4);

    // 1) conversions
    cvt_f32_to_bf16_kernel<<<4096, 256, 0, stream>>>(fq, queue_b, (size_t)Q * D);
    cvt_f32_to_bf16_kernel<<<512, 256, 0, stream>>>(h1, h_b, (size_t)Bsz * D);
    cvt_f32_to_bf16_kernel<<<512, 256, 0, stream>>>(h2, h_b + (size_t)Bsz * D, (size_t)Bsz * D);
    transpose_cvt_kernel<<<D, D, 0, stream>>>(W1, w1t, D);
    transpose_cvt_kernel<<<D, D, 0, stream>>>(W2, w2t, D);

    // 2) projection: y = relu(h @ W1 + b1)  -> bf16
    gemm_bf16_kernel<<<dim3(M2 / 16, D / 128), 256, 0, stream>>>(
        h_b, w1t, b1, nullptr, y_b, D, 1.0f, 1);
    //    z = y @ W2 + b2 -> f32
    gemm_bf16_kernel<<<dim3(M2 / 16, D / 128), 256, 0, stream>>>(
        y_b, w2t, b2, z_f, nullptr, D, 1.0f, 0);

    // 3) L2 normalize
    l2norm_kernel<<<M2, 256, 0, stream>>>(z_f, zn_f, zn_b);

    // 4) fused NN search (argmax over 65536 queue rows), then gather nn rows
    nn_search_kernel<<<M2 / 32, 256, 0, stream>>>(zn_b, queue_b, nn_idx, Q);
    gather_rows_kernel<<<M2, 256, 0, stream>>>(queue_b, nn_idx, nn_b);

    // 5) logits L1 = nn1 @ z2n^T / TEMP ; CE over rows and columns
    gemm_bf16_kernel<<<dim3(Bsz / 16, Bsz / 128), 256, 0, stream>>>(
        nn_b, zn_b + (size_t)Bsz * D, nullptr, L, nullptr, Bsz, INV_TEMP, 0);
    row_lse_kernel<<<Bsz, 256, 0, stream>>>(L, Bsz, lr1, dg1);
    col_lse_partial_kernel<<<dim3(Bsz / 256, 8), 256, 0, stream>>>(L, Bsz, Bsz / 8, pm, ps);
    col_lse_combine_kernel<<<Bsz / 256, 256, 0, stream>>>(pm, ps, Bsz, 8, lc1);

    // 6) logits L2 = nn2 @ z1n^T / TEMP (reuse L buffer)
    gemm_bf16_kernel<<<dim3(Bsz / 16, Bsz / 128), 256, 0, stream>>>(
        nn_b + (size_t)Bsz * D, zn_b, nullptr, L, nullptr, Bsz, INV_TEMP, 0);
    row_lse_kernel<<<Bsz, 256, 0, stream>>>(L, Bsz, lr2, dg2);
    col_lse_partial_kernel<<<dim3(Bsz / 256, 8), 256, 0, stream>>>(L, Bsz, Bsz / 8, pm, ps);
    col_lse_combine_kernel<<<Bsz / 256, 256, 0, stream>>>(pm, ps, Bsz, 8, lc2);

    // 7) loss scalar
    final_loss_kernel<<<1, 256, 0, stream>>>(lr1, lc1, dg1, lr2, lc2, dg2, Bsz, out);

    // 8) FIFO queue update: new_queue = [z1n ; feature_queue[:-B]]
    copy_f32_kernel<<<2048, 256, 0, stream>>>(zn_f, out + 1, (size_t)Bsz * D);
    copy_f32_kernel<<<8192, 256, 0, stream>>>(fq, out + 1 + (size_t)Bsz * D,
                                              (size_t)(Q - Bsz) * D);
}
